// MLP_2095944040849
// MI455X (gfx1250) — matrix operand order repaired and checked by reading, not compiled, not run on hardware
//
#include <hip/hip_runtime.h>

typedef __attribute__((ext_vector_type(16))) _Float16 v16h;
typedef __attribute__((ext_vector_type(8)))  _Float16 v8h;
typedef __attribute__((ext_vector_type(4)))  _Float16 v4h;
typedef __attribute__((ext_vector_type(8)))  float    v8f;
typedef __attribute__((ext_vector_type(4)))  float    v4f;

#define NN   512      // N = OUT = 512
#define MT   128      // rows of x per block
#define XP   520      // LDS pitch (halves) for 512-wide tiles (bank-conflict pad)
#define WP   40       // LDS pitch (halves) for staged 32-wide weight chunks
#define NCW  256      // output-column chunk width (2 chunks cover 512)

#if __has_builtin(__builtin_amdgcn_global_load_async_to_lds_b128)
#define ASYNC_LDS 1
#else
#define ASYNC_LDS 0
#endif

#if ASYNC_LDS
// Signature discovered via round-2/3 diagnostics:
//   (AS(1) int4* src, AS(3) int4* dst, imm int offset, imm int cpol)
typedef int v4i_g __attribute__((vector_size(16)));
typedef __attribute__((address_space(1))) v4i_g* as1_v4i;
typedef __attribute__((address_space(3))) v4i_g* as3_v4i;
#endif

// ---------------------------------------------------------------------------
// Kernel 1: W reconstruction from Krylov structure, stored TRANSPOSED in f16.
// W[a,b] = sum_j sigma(b,j) * dot4(G[(a-j)%512,:], H[(b+j)%512,:])
//   sigma(b,j) = -1 iff b+j >= 512 (skew shift wraps at most once for j<512)
// Store WT[b*512 + a] = W[a][b]  (so GEMM B-operand is contiguous in K=a).
// ---------------------------------------------------------------------------
__global__ __launch_bounds__(256) void build_wt(const float* __restrict__ G,
                                                const float* __restrict__ H,
                                                _Float16* __restrict__ WT) {
  __shared__ v4f sG[NN];
  __shared__ v4f sH4[NN];
  const int a = blockIdx.x;
  for (int u = threadIdx.x; u < NN; u += 256) {
    sG[u]  = *(const v4f*)(G + u * 4);
    sH4[u] = *(const v4f*)(H + u * 4);
  }
  __syncthreads();
  for (int b = threadIdx.x; b < NN; b += 256) {
    float acc = 0.f;
    #pragma unroll 4
    for (int j = 0; j < NN; ++j) {
      const int ig = (a - j) & (NN - 1);
      const int ih = (b + j) & (NN - 1);
      v4f g = sG[ig];
      v4f h = sH4[ih];
      float d = g[0] * h[0] + g[1] * h[1] + g[2] * h[2] + g[3] * h[3];
      acc += (b + j >= NN) ? -d : d;
    }
    WT[b * NN + a] = (_Float16)acc;
  }
}

// ---------------------------------------------------------------------------
// Kernel 2: W2 (512x512 f32, row-major k x o) -> W2T (o x k, f16)
// ---------------------------------------------------------------------------
__global__ __launch_bounds__(256) void build_w2t(const float* __restrict__ W2,
                                                 _Float16* __restrict__ W2T) {
  const int o = blockIdx.x;
  for (int k = threadIdx.x; k < NN; k += 256)
    W2T[o * NN + k] = (_Float16)W2[k * NN + o];
}

// ---------------------------------------------------------------------------
// Stage one 256x32 weight chunk row (64B per thread) into an sW buffer.
// ---------------------------------------------------------------------------
__device__ __forceinline__ void stage_w(const _Float16* __restrict__ Bsrc,
                                        int n0, int k, _Float16* dstbase,
                                        int tid) {
  const _Float16* src = Bsrc + (size_t)(n0 + tid) * NN + k;
  _Float16* dst = dstbase + tid * WP;
#if ASYNC_LDS
  __builtin_amdgcn_global_load_async_to_lds_b128((as1_v4i)src, (as3_v4i)dst, 0, 0);
  __builtin_amdgcn_global_load_async_to_lds_b128((as1_v4i)src, (as3_v4i)dst, 16, 0);
  __builtin_amdgcn_global_load_async_to_lds_b128((as1_v4i)src, (as3_v4i)dst, 32, 0);
  __builtin_amdgcn_global_load_async_to_lds_b128((as1_v4i)src, (as3_v4i)dst, 48, 0);
#else
  v8h w0 = *(const v8h*)(src);
  v8h w1 = *(const v8h*)(src + 8);
  v8h w2 = *(const v8h*)(src + 16);
  v8h w3 = *(const v8h*)(src + 24);
  *(v8h*)(dst)      = w0;
  *(v8h*)(dst + 8)  = w1;
  *(v8h*)(dst + 16) = w2;
  *(v8h*)(dst + 24) = w3;
#endif
}

__device__ __forceinline__ void wait_async() {
#if ASYNC_LDS
#if __has_builtin(__builtin_amdgcn_s_wait_asynccnt)
  __builtin_amdgcn_s_wait_asynccnt(0);
#else
  asm volatile("s_wait_asynccnt 0" ::: "memory");
#endif
#endif
}

// ---------------------------------------------------------------------------
// Kernel 3: fused  out = relu(x@W + b1) @ W2 + b2
// Block: 256 threads (8 waves as a 2x4 grid), 128 rows of x.
// Per wave: 64x64 tile = 4 m-tiles x 4 n-tiles = 16 WMMA accumulators.
// Weight chunks double-buffered in LDS via async global->LDS; the async
// issue for chunk k+1 overlaps the 16-WMMA compute on chunk k, with a single
// workgroup barrier per K-step.
// ---------------------------------------------------------------------------
__global__ __launch_bounds__(256) void fused_mlp(const float* __restrict__ x,
                                                 const float* __restrict__ b1,
                                                 const float* __restrict__ b2,
                                                 const _Float16* __restrict__ WT,
                                                 const _Float16* __restrict__ W2T,
                                                 float* __restrict__ out) {
  __shared__ _Float16 sX[MT * XP];     // x tile, f16            (133120 B)
  __shared__ _Float16 sHt[MT * XP];    // relu(h) tile, f16      (133120 B)
  __shared__ _Float16 sW[2][NCW * WP]; // weight chunks, 2 bufs  (40960 B)

  const int tid  = threadIdx.x;
  const int lane = tid & 31;
  const int wave = tid >> 5;
  const int mg   = wave >> 2;        // 0..1 -> rows 64*mg .. +63
  const int ng   = wave & 3;         // 0..3 -> cols 64*ng .. +63 (within chunk)
  const int lrow = lane & 15;
  const int lhi  = lane >> 4;        // 0|1
  const long row0 = (long)blockIdx.x * MT;

  // ---- load x tile (128x512 f32) -> f16 into sX ----
  {
    const float* xb = x + row0 * NN;
    #pragma unroll
    for (int i = 0; i < 64; ++i) {
      const int idx = tid + i * 256;      // 0..16383
      const int r = idx >> 7;             // row 0..127
      const int c = (idx & 127) << 2;     // col (multiple of 4)
      v4f v = *(const v4f*)(xb + r * NN + c);
      v4h hv;
      hv[0] = (_Float16)v[0]; hv[1] = (_Float16)v[1];
      hv[2] = (_Float16)v[2]; hv[3] = (_Float16)v[3];
      *(v4h*)(&sX[r * XP + c]) = hv;
    }
  }
  // (the barrier at the top of the first K-step covers sX visibility)

  for (int phase = 0; phase < 2; ++phase) {
    const _Float16* __restrict__ Bsrc = (phase == 0) ? WT : W2T;
    const _Float16* Asrc              = (phase == 0) ? sX : sHt;

    for (int nc = 0; nc < 2; ++nc) {
      const int n0 = nc * NCW;
      v8f acc[4][4] = {};

      stage_w(Bsrc, n0, 0, sW[0], tid);  // prologue: fill buffer 0

      for (int ks = 0; ks < 16; ++ks) {
        const int k   = ks * 32;
        const int cur = ks & 1;

        wait_async();     // our 4 async writes for sW[cur] are done
        __syncthreads();  // all waves' writes visible; prev readers finished

        if (ks < 15)      // overlap next stage with this K-step's compute
          stage_w(Bsrc, n0, k + 32, sW[cur ^ 1], tid);

        const _Float16* sWc = sW[cur];

        // A fragments: lane = row (l%16); K halves 0-7 at k+8*lhi, 8-15 at +16
        v16h afrag[4];
        #pragma unroll
        for (int mi = 0; mi < 4; ++mi) {
          const _Float16* arow =
              Asrc + (64 * mg + 16 * mi + lrow) * XP + k + lhi * 8;
          v8h a_lo = *(const v8h*)(arow);
          v8h a_hi = *(const v8h*)(arow + 16);
          afrag[mi] = __builtin_shufflevector(a_lo, a_hi, 0, 1, 2, 3, 4, 5, 6,
                                              7, 8, 9, 10, 11, 12, 13, 14, 15);
        }
        #pragma unroll
        for (int ni = 0; ni < 4; ++ni) {
          // B fragment: lane = col (l%16); K halves 0-7 at 8*lhi, 8-15 at +16
          const int ncol = 64 * ng + 16 * ni + lrow;
          const _Float16* brow = &sWc[ncol * WP + lhi * 8];
          v8h b_lo = *(const v8h*)(brow);
          v8h b_hi = *(const v8h*)(brow + 16);
          v16h bfrag = __builtin_shufflevector(b_lo, b_hi, 0, 1, 2, 3, 4, 5, 6,
                                               7, 8, 9, 10, 11, 12, 13, 14, 15);
          #pragma unroll
          for (int mi = 0; mi < 4; ++mi) {
            acc[mi][ni] = __builtin_amdgcn_wmma_f32_16x16x32_f16(
                false, afrag[mi], false, bfrag, (short)0, acc[mi][ni], false,
                false);
          }
        }
      }

      // ---- epilogue: C layout = (row v + 8*lhi, col lane%16) ----
      if (phase == 0) {
        #pragma unroll
        for (int mi = 0; mi < 4; ++mi) {
          #pragma unroll
          for (int ni = 0; ni < 4; ++ni) {
            const int col = n0 + 64 * ng + 16 * ni + lrow;
            const float bias = b1[col];
            #pragma unroll
            for (int v = 0; v < 8; ++v) {
              float val = acc[mi][ni][v] + bias;
              val = val > 0.f ? val : 0.f;
              const int m = 64 * mg + 16 * mi + v + 8 * lhi;
              sHt[m * XP + col] = (_Float16)val;
            }
          }
        }
      } else {
        #pragma unroll
        for (int mi = 0; mi < 4; ++mi) {
          #pragma unroll
          for (int ni = 0; ni < 4; ++ni) {
            const int col = n0 + 64 * ng + 16 * ni + lrow;
            const float bias = b2[col];
            #pragma unroll
            for (int v = 0; v < 8; ++v) {
              const int m = 64 * mg + 16 * mi + v + 8 * lhi;
              out[(row0 + m) * NN + col] = acc[mi][ni][v] + bias;
            }
          }
        }
      }
    }
  }
}

// ---------------------------------------------------------------------------
extern "C" void kernel_launch(void* const* d_in, const int* in_sizes, int n_in,
                              void* d_out, int out_size, void* d_ws,
                              size_t ws_size, hipStream_t stream) {
  const float* x  = (const float*)d_in[0];
  const float* G  = (const float*)d_in[1];
  const float* H  = (const float*)d_in[2];
  const float* b1 = (const float*)d_in[3];
  const float* W2 = (const float*)d_in[4];
  const float* b2 = (const float*)d_in[5];

  _Float16* WT  = (_Float16*)d_ws;          // 512*512*2 = 512 KB
  _Float16* W2T = WT + NN * NN;             // +512 KB

  build_wt<<<NN, 256, 0, stream>>>(G, H, WT);
  build_w2t<<<NN, 256, 0, stream>>>(W2, W2T);

  const int nblocks = 65536 / MT;  // 512
  fused_mlp<<<nblocks, 256, 0, stream>>>(x, b1, b2, WT, W2T, (float*)d_out);
}
